// IGNN_21019569947057
// MI455X (gfx1250) — matrix operand order repaired and checked by reading, not compiled, not run on hardware
//
#include <hip/hip_runtime.h>

#define HID   64
#define DIN   128
#define DOUT  40
#define KAPPA 0.9f
#define POWER_ITERS 50
#define PROPS 300

typedef __attribute__((ext_vector_type(16))) __bf16 v16bf;
typedef __attribute__((ext_vector_type(8)))  float  v8f;

__device__ __forceinline__ void atomAddF(float* p, float v) { unsafeAtomicAdd(p, v); }

// ---------------- setup kernels ----------------
__global__ void k_init(int* deg, float* scal, int n){
  int i = blockIdx.x*blockDim.x + threadIdx.x;
  if (i < n) deg[i] = 1;             // self-loop weight contributes 1 to deg
  if (i < 8) scal[i] = 0.f;
}

__global__ void k_deg(const int* src, const int* dst, int* deg, int e){
  int i = blockIdx.x*blockDim.x + threadIdx.x;
  if (i >= e) return;
  int s = src[i], d = dst[i];
  if (s != d) atomicAdd(&deg[s], 1);
}

__global__ void k_dinv(const int* deg, float* dinv, int n){
  int i = blockIdx.x*blockDim.x + threadIdx.x;
  if (i < n) dinv[i] = rsqrtf((float)deg[i]);
}

__global__ void k_build(const int* src, const int* dst, const float* dinv,
                        int* rows, int* cols, float* vals, float* pv, int n, int nnz){
  int i = blockIdx.x*blockDim.x + threadIdx.x;
  if (i >= nnz) return;
  if (i < n) {
    rows[i] = i; cols[i] = i; vals[i] = dinv[i]*dinv[i];
    pv[i] = 1.f;                                   // power-iteration v0 = ones
  } else {
    int e = i - n;
    int s = src[e], d = dst[e];
    rows[i] = s; cols[i] = d;
    vals[i] = (s != d) ? dinv[s]*dinv[d] : 0.f;    // vals >= 0, so |adj| == adj
  }
}

// ---------------- power iteration (spectral radius) ----------------
__global__ void k_zero_u(float* u, float* scal, int n){
  int i = blockIdx.x*blockDim.x + threadIdx.x;
  if (i < n) u[i] = 0.f;
  if (i == 0) scal[0] = 0.f;
}

__global__ void k_spmv(const int* __restrict__ rows, const int* __restrict__ cols,
                       const float* __restrict__ vals, const float* __restrict__ v,
                       float* __restrict__ u, int nnz){
  int i = blockIdx.x*blockDim.x + threadIdx.x;
  if (i >= nnz) return;
  float a = vals[i];
  if (a != 0.f) atomAddF(&u[rows[i]], a * v[cols[i]]);
}

__global__ void k_sumsq(const float* __restrict__ u, float* scal, int n){
  __shared__ float red[256];
  int i = blockIdx.x*256 + threadIdx.x;
  float x = (i < n) ? u[i] : 0.f;
  red[threadIdx.x] = x*x;
  __syncthreads();
  for (int s = 128; s > 0; s >>= 1){
    if (threadIdx.x < s) red[threadIdx.x] += red[threadIdx.x + s];
    __syncthreads();
  }
  if (threadIdx.x == 0) atomAddF(&scal[0], red[0]);
}

__global__ void k_vnorm(const float* u, const float* scal, float* v, int n){
  int i = blockIdx.x*blockDim.x + threadIdx.x;
  if (i >= n) return;
  float nrm = fmaxf(sqrtf(scal[0]), 1e-30f);
  v[i] = u[i] / nrm;
}

__global__ void k_rad(float* scal){
  float rad = sqrtf(scal[0]) + 1e-5f;
  scal[1] = rad;
  scal[2] = KAPPA / rad;   // effective kappa for L1 projection
}

// ---------------- per-column L1-ball projection of igc_w ----------------
__global__ void k_proj(const float* __restrict__ W, const float* scal, float* __restrict__ Wp){
  int c = threadIdx.x;
  if (c >= HID) return;
  float kap = scal[2];
  float a[HID]; float colsum = 0.f;
  for (int i = 0; i < HID; i++){ float t = fabsf(W[i*HID + c]); a[i] = t; colsum += t; }
  for (int i = 1; i < HID; i++){                 // insertion sort, descending
    float key = a[i]; int j = i - 1;
    while (j >= 0 && a[j] < key){ a[j+1] = a[j]; j--; }
    a[j+1] = key;
  }
  int rho = 0; float css = 0.f;
  for (int j = 0; j < HID; j++){ css += a[j]; if (a[j]*(float)(j+1) - css + kap > 0.f) rho++; }
  if (rho < 1) rho = 1;
  css = 0.f;
  for (int j = 0; j < rho; j++) css += a[j];
  float theta = (css - kap) / (float)rho;
  theta = (colsum > kap) ? fmaxf(theta, 0.f) : 0.f;
  for (int i = 0; i < HID; i++){
    float w = W[i*HID + c];
    float m = fmaxf(fabsf(w) - theta, 0.f);
    Wp[i*HID + c] = (w < 0.f) ? -m : m;          // w==0 -> m==0
  }
}

// ---------------- pack B (K x 64, row-major f32) into WMMA bf16 fragments ----
// Layout per (ntile t, kstep s): lane l holds B column n = t*16+(l&15), element i
// holds k = s*32 + (i<8 ? 8*(l>>4)+i : 16+8*(l>>4)+(i-8)) (table order, matches A).
__global__ void k_make_frags(const float* __restrict__ W, unsigned short* __restrict__ frag, int ksteps){
  int tid = blockIdx.x*blockDim.x + threadIdx.x;
  if (tid >= 4*ksteps*32) return;
  int lane = tid & 31;
  int s = (tid >> 5) % ksteps;
  int t = tid / (32*ksteps);
  const float* src = W + (size_t)(s*32)*HID + t*16 + (lane & 15);
  unsigned short* dst = frag + (size_t)tid*16;   // tid == ((t*ksteps+s)*32+lane)
  #pragma unroll
  for (int i = 0; i < 16; i++){
    __bf16 b = (__bf16)src[(size_t)(i < 8 ? 8*(lane >> 4) + i : 16 + 8*(lane >> 4) + (i - 8))*HID];
    dst[i] = __builtin_bit_cast(unsigned short, b);
  }
}

// ---------------- WMMA GEMM: og = [relu](A) @ B + bvec ; oc = csrc ? csrc : og
// A: nrows x (32*KSTEPS) f32 row-major. Output nrows x 64. One wave per 16 rows.
template<int KSTEPS, bool RELU_A>
__global__ __launch_bounds__(128) void k_gemm(
    const float* __restrict__ A, const unsigned short* __restrict__ bfrag,
    const float* __restrict__ bvec, const float* __restrict__ csrc,
    float* __restrict__ og, float* __restrict__ oc, int nrows)
{
  constexpr int KW  = KSTEPS*32;
  constexpr int LST = KW + 4;                // pad to spread LDS banks
  constexpr int NW  = 4;                     // 4 waves / block (wave32)
  __shared__ float sl[NW*16*LST];
  const int lane = threadIdx.x & 31;
  const int wv   = threadIdx.x >> 5;
  const int r0   = (blockIdx.x*NW + wv)*16;
  float* sh = sl + wv*16*LST;

  // stage 16 x KW A-tile (contiguous in memory) into LDS, coalesced float4
  const bool full = (r0 + 16) <= nrows;
  for (int i = lane; i < 16*KW/4; i += 32){
    const int row = (i*4)/KW, col = (i*4)%KW;
    const int rr  = r0 + row;
    float4 v = make_float4(0.f, 0.f, 0.f, 0.f);
    if (full || rr < nrows) v = *(const float4*)(A + (size_t)rr*KW + col);
    sh[row*LST+col] = v.x; sh[row*LST+col+1] = v.y;
    sh[row*LST+col+2] = v.z; sh[row*LST+col+3] = v.w;
  }
  // wave-private LDS slice; same-wave DS ops complete in order -> no barrier.

  const int m = lane & 15, half = lane >> 4;
  v8f acc[4] = {};
  #pragma unroll
  for (int ks = 0; ks < KSTEPS; ks++){
    // A fragment: 16-bit A 16x32 layout — VGPR j holds K pair klo,klo+1
    v16bf a;
    #pragma unroll
    for (int j = 0; j < 8; j++){
      const int klo = ks*32 + (j < 4 ? 8*half + 2*j : 16 + 8*half + 2*(j-4));
      float x0 = sh[m*LST + klo], x1 = sh[m*LST + klo + 1];
      if (RELU_A){ x0 = fmaxf(x0, 0.f); x1 = fmaxf(x1, 0.f); }
      a[2*j]   = (__bf16)x0;
      a[2*j+1] = (__bf16)x1;
    }
    #pragma unroll
    for (int t = 0; t < 4; t++){
      v16bf b = *(const v16bf*)(bfrag + ((size_t)(t*KSTEPS + ks)*32 + lane)*16);
      acc[t] = __builtin_amdgcn_wmma_f32_16x16x32_bf16(
          false, a, false, b, (short)0, acc[t], false, false);
    }
  }

  // C/D layout: lane -> n = lane&15, VGPR j -> row m = j + 8*half
  const int n = lane & 15;
  #pragma unroll
  for (int t = 0; t < 4; t++){
    #pragma unroll
    for (int j = 0; j < 8; j++){
      const int row = r0 + j + 8*half;
      if (row < nrows){
        const size_t idx = (size_t)row*HID + t*16 + n;
        float dv = acc[t][j];
        if (bvec) dv += bvec[t*16 + n];
        og[idx] = dv;
        if (oc) oc[idx] = csrc ? csrc[idx] : dv;  // fused h_next = bias (or = result)
      }
    }
  }
}

// ---------------- edge scatter: h[rows] += vals * zw[cols]  (8 feats/thread) ---
__global__ void k_scatter(const int* __restrict__ rows, const int* __restrict__ cols,
                          const float* __restrict__ vals, const float* __restrict__ zw,
                          float* __restrict__ h, int nnz){
  int tid = blockIdx.x*blockDim.x + threadIdx.x;
  int e = tid >> 3;
  if (e >= nnz) return;
  float a = vals[e];
  if (a == 0.f) return;
  int g = (tid & 7)*8;
  const float4* zp = (const float4*)(zw + (size_t)cols[e]*HID + g);
  float4 z0 = zp[0], z1 = zp[1];
  float* hp = h + (size_t)rows[e]*HID + g;
  atomAddF(hp+0, a*z0.x); atomAddF(hp+1, a*z0.y);
  atomAddF(hp+2, a*z0.z); atomAddF(hp+3, a*z0.w);
  atomAddF(hp+4, a*z1.x); atomAddF(hp+5, a*z1.y);
  atomAddF(hp+6, a*z1.z); atomAddF(hp+7, a*z1.w);
}

// ---------------- final: z = relu(h); row-normalize; out = zn @ w_cls + b_cls --
__global__ __launch_bounds__(256) void k_final(const float* __restrict__ h,
    const float* __restrict__ wcls, const float* __restrict__ bcls,
    float* __restrict__ out, int n){
  __shared__ float zs[8][HID];
  int wv = threadIdx.x >> 5, lane = threadIdx.x & 31;   // one wave32 per row
  int row = blockIdx.x*8 + wv;
  float z0 = 0.f, z1 = 0.f;
  if (row < n){
    z0 = fmaxf(h[(size_t)row*HID + 2*lane],     0.f);
    z1 = fmaxf(h[(size_t)row*HID + 2*lane + 1], 0.f);
  }
  float ss = z0*z0 + z1*z1;
  #pragma unroll
  for (int off = 16; off > 0; off >>= 1) ss += __shfl_xor(ss, off, 32);
  float inv = 1.f / fmaxf(sqrtf(ss), 1e-12f);
  zs[wv][2*lane] = z0*inv; zs[wv][2*lane+1] = z1*inv;
  __syncthreads();
  if (row < n){
    for (int o = lane; o < DOUT; o += 32){
      float acc = bcls[o];
      #pragma unroll
      for (int k = 0; k < HID; k++) acc += zs[wv][k]*wcls[k*DOUT + o];
      out[(size_t)row*DOUT + o] = acc;
    }
  }
}

// ---------------- host driver (graph-capture safe: launches only) ----------------
extern "C" void kernel_launch(void* const* d_in, const int* in_sizes, int n_in,
                              void* d_out, int out_size, void* d_ws, size_t ws_size,
                              hipStream_t stream){
  const float* x     = (const float*)d_in[0];
  const int*   edges = (const int*)  d_in[1];
  const float* w_gnn = (const float*)d_in[2];
  const float* b_gnn = (const float*)d_in[3];
  const float* igc_w = (const float*)d_in[4];
  const float* w_cls = (const float*)d_in[5];
  const float* b_cls = (const float*)d_in[6];
  float* out = (float*)d_out;

  const int N   = in_sizes[0] / DIN;
  const int E   = in_sizes[1] / 2;
  const int NNZ = N + E;
  const int* src = edges;
  const int* dst = edges + E;

  char* wp = (char*)d_ws;
  auto take = [&](size_t bytes) -> char* {
    char* p = wp; wp += (bytes + 255) & ~(size_t)255; return p;
  };
  int*   rows  = (int*)  take((size_t)NNZ*4);
  int*   cols  = (int*)  take((size_t)NNZ*4);
  float* vals  = (float*)take((size_t)NNZ*4);
  int*   deg   = (int*)  take((size_t)N*4);
  float* dinv  = (float*)take((size_t)N*4);
  float* pv    = (float*)take((size_t)N*4);
  float* pu    = (float*)take((size_t)N*4);
  float* scal  = (float*)take(64*4);
  float* wproj = (float*)take((size_t)HID*HID*4);
  unsigned short* wfrag  = (unsigned short*)take((size_t)4*2*32*16*2);
  unsigned short* wgfrag = (unsigned short*)take((size_t)4*4*32*16*2);
  float* bias  = (float*)take((size_t)N*HID*4);
  float* hA    = (float*)take((size_t)N*HID*4);
  float* hB    = (float*)take((size_t)N*HID*4);
  float* zw    = (float*)take((size_t)N*HID*4);

  dim3 blk(256);
  auto g1 = [&](long total){ return dim3((unsigned)((total + 255)/256)); };

  // --- adjacency build ---
  k_init <<<g1(N),   blk, 0, stream>>>(deg, scal, N);
  k_deg  <<<g1(E),   blk, 0, stream>>>(src, dst, deg, E);
  k_dinv <<<g1(N),   blk, 0, stream>>>(deg, dinv, N);
  k_build<<<g1(NNZ), blk, 0, stream>>>(src, dst, dinv, rows, cols, vals, pv, N, NNZ);

  // --- spectral radius via power iteration ---
  for (int it = 0; it < POWER_ITERS; it++){
    k_zero_u<<<g1(N),   blk, 0, stream>>>(pu, scal, N);
    k_spmv  <<<g1(NNZ), blk, 0, stream>>>(rows, cols, vals, pv, pu, NNZ);
    k_sumsq <<<g1(N),   blk, 0, stream>>>(pu, scal, N);
    k_vnorm <<<g1(N),   blk, 0, stream>>>(pu, scal, pv, N);
  }
  k_zero_u<<<g1(N),   blk, 0, stream>>>(pu, scal, N);
  k_spmv  <<<g1(NNZ), blk, 0, stream>>>(rows, cols, vals, pv, pu, NNZ);
  k_sumsq <<<g1(N),   blk, 0, stream>>>(pu, scal, N);
  k_rad   <<<dim3(1), dim3(1), 0, stream>>>(scal);

  // --- projection + weight packing ---
  k_proj<<<dim3(1), dim3(64), 0, stream>>>(igc_w, scal, wproj);
  k_make_frags<<<g1(4*2*32), blk, 0, stream>>>(wproj, wfrag, 2);
  k_make_frags<<<g1(4*4*32), blk, 0, stream>>>(w_gnn, wgfrag, 4);

  // --- bias = x @ w_gnn + b_gnn ; h1 = bias (fused second store) ---
  const int RPB = 4*16;  // rows per block (4 waves x 16)
  dim3 gg((unsigned)((N + RPB - 1)/RPB)), gb(128);
  k_gemm<4,false><<<gg, gb, 0, stream>>>(x, wgfrag, b_gnn, nullptr, bias, hA, N);

  // --- 300 x { zw = relu(h)@Wp (+ h_next=bias fused) ; scatter-add edges } ---
  float* hc = hA; float* hn = hB;
  for (int it = 0; it < PROPS; it++){
    k_gemm<2,true><<<gg, gb, 0, stream>>>(hc, wfrag, nullptr, bias, zw, hn, N);
    k_scatter<<<g1((long)NNZ*8), blk, 0, stream>>>(rows, cols, vals, zw, hn, NNZ);
    float* t = hc; hc = hn; hn = t;
  }

  // --- final relu + row-normalize + classifier ---
  k_final<<<dim3((unsigned)((N + 7)/8)), dim3(256), 0, stream>>>(hc, w_cls, b_cls, out, N);

  (void)n_in; (void)out_size; (void)ws_size;
}
